// SSM_13984413516253
// MI455X (gfx1250) — matrix operand order repaired and checked by reading, not compiled, not run on hardware
//
#include <hip/hip_runtime.h>
#include <cstdint>
#include <cstddef>

// ---------------------------------------------------------------------------
// Problem constants: CPD=256, IDIM=1024, SDIM=1024, BATCH=128, FRAMES=512.
// ---------------------------------------------------------------------------
#define BATCH  128
#define CPD    256
#define FRAMES 512
#define IDIM   1024
#define SDIM   1024
#define HOP    512

typedef __bf16 bf16_t;
typedef __attribute__((ext_vector_type(16))) __bf16 v16bf;
typedef __attribute__((ext_vector_type(8)))  __bf16 v8bf;
typedef __attribute__((ext_vector_type(8)))  float  v8f;
typedef __attribute__((ext_vector_type(4)))  unsigned int v4u;
typedef __attribute__((ext_vector_type(8)))  int v8i;
typedef __attribute__((ext_vector_type(4)))  int v4i;

// ---------------------------------------------------------------------------
// CDNA5 async global->LDS copy (ASYNCcnt tracked) + wait
// ---------------------------------------------------------------------------
__device__ __forceinline__ void async_b128(void* lds_dst, const void* gsrc) {
  unsigned l = (unsigned)(uintptr_t)lds_dst;
  unsigned long long g = (unsigned long long)(uintptr_t)gsrc;
  asm volatile("global_load_async_to_lds_b128 %0, %1, off"
               :: "v"(l), "v"(g) : "memory");
}
#define WAIT_ASYNC0() asm volatile("s_wait_asynccnt 0" ::: "memory")

// ---------------------------------------------------------------------------
// CDNA5 Tensor Data Mover: 2D tile (rows x cols bf16, strided rows) -> LDS.
// D# built per ISA §8.3/8.4: group0 {count=1, lds_addr, global_addr, type=2},
// group1 {data_size=2B, tensor_dim0=cols, tensor_dim1=65536, tile_dim0=cols,
//          tile_dim1=rows, tensor_dim0_stride=row_stride}. Groups 2/3 zero
// (2D tensor). Tracked by TENSORcnt.
// ---------------------------------------------------------------------------
__device__ __forceinline__ void tdm_load_2d_bf16(unsigned lds_addr,
                                                 const void* gptr,
                                                 unsigned tile_rows,
                                                 unsigned tile_cols,
                                                 unsigned long long row_stride) {
  unsigned long long ga = (unsigned long long)(uintptr_t)gptr;
  v4u g0;
  g0[0] = 1u;                                          // count=1 (valid D#)
  g0[1] = lds_addr;                                    // LDS byte address
  g0[2] = (unsigned)(ga & 0xFFFFFFFFu);                // global_addr[31:0]
  g0[3] = (unsigned)((ga >> 32) & 0x01FFFFFFu) | 0x80000000u;  // [56:32]|type=2
  v8i g1;
  g1[0] = 0x00010000;                                  // data_size=1 (2 bytes)
  g1[1] = (int)((tile_cols & 0xFFFFu) << 16);          // tensor_dim0[15:0]
  g1[2] = (int)((tile_cols >> 16) & 0xFFFFu);          // tensor_dim0[31:16]
  g1[3] = (int)(1u | ((tile_cols & 0xFFFFu) << 16));   // tensor_dim1=65536; tile_dim0
  g1[4] = (int)(tile_rows & 0xFFFFu);                  // tile_dim1; tile_dim2=0
  g1[5] = (int)(unsigned)(row_stride & 0xFFFFFFFFu);   // dim0_stride[31:0]
  g1[6] = (int)(unsigned)((row_stride >> 32) & 0xFFFFu);
  g1[7] = 0;
  v4i z4 = {0, 0, 0, 0};
#if defined(__clang_major__) && (__clang_major__ >= 23)
  v8i z8 = {0, 0, 0, 0, 0, 0, 0, 0};
  __builtin_amdgcn_tensor_load_to_lds(g0, g1, z4, z4, z8, 0);
#else
  __builtin_amdgcn_tensor_load_to_lds(g0, g1, z4, z4, 0);
#endif
}
#define WAIT_TENSOR0() __builtin_amdgcn_s_wait_tensorcnt(0)

// ---------------------------------------------------------------------------
// WMMA helpers (wave32; 16x16x32 bf16 -> f32)
// ---------------------------------------------------------------------------
__device__ __forceinline__ v8f wmma_bf(v16bf a, v16bf b, v8f c) {
  return __builtin_amdgcn_wmma_f32_16x16x32_bf16(
      false, a, false, b, (short)0, c, false, false);
}

__device__ __forceinline__ v16bf cat8(v8bf lo, v8bf hi) {
  return __builtin_shufflevector(lo, hi, 0, 1, 2, 3, 4, 5, 6, 7,
                                 8, 9, 10, 11, 12, 13, 14, 15);
}

__device__ __forceinline__ v16bf load_frag16(const bf16_t* p) {
  const v8bf* q = (const v8bf*)p;
  return cat8(q[0], q[1]);
}

// B fragment from column-major BT[n][k]: K = hf*8+0..7, then 16+hf*8+0..7.
__device__ __forceinline__ v16bf load_b_col(const bf16_t* BT, int kstride,
                                            int n0, int k0, int lane) {
  int n = n0 + (lane & 15);
  int hf = lane >> 4;
  const bf16_t* p = BT + (size_t)n * kstride + k0 + hf * 8;
  return cat8(*(const v8bf*)p, *(const v8bf*)(p + 16));
}

// A fragment from a row-major row pointer: two b128 loads.
__device__ __forceinline__ v16bf load_a_row(const bf16_t* row, int k0, int hf) {
  const v8bf* q0 = (const v8bf*)(row + k0 + hf * 8);
  const v8bf* q1 = (const v8bf*)(row + k0 + 16 + hf * 8);
  return cat8(*q0, *q1);
}

// ---------------------------------------------------------------------------
// 1) Transpose f32 -> bf16 col-major:  out[c*R + r] = (bf16) in[r*C + c]
// ---------------------------------------------------------------------------
__global__ void transpose_to_bf16(const float* __restrict__ in,
                                  bf16_t* __restrict__ out, int R, int C) {
  int c = blockIdx.x * 16 + threadIdx.x;
  int r = blockIdx.y * 16 + threadIdx.y;
  if (r < R && c < C) out[(size_t)c * R + r] = (bf16_t)in[(size_t)r * C + c];
}

// ---------------------------------------------------------------------------
// 2) x[b,f,i] = sum_c control[b,c,f] * proj[c,i]   (M=F, N=IDIM, K=CPD)
// ---------------------------------------------------------------------------
__global__ __launch_bounds__(128) void gemm_x_kernel(
    const float* __restrict__ control, const bf16_t* __restrict__ projT,
    bf16_t* __restrict__ x) {
  int lane = threadIdx.x & 31, w = threadIdx.x >> 5;
  int hf = lane >> 4, m16 = lane & 15;
  int n0 = blockIdx.x * 128 + w * 32;   // two tiles: n0, n0+16
  int f0 = blockIdx.y * 16;
  int b  = blockIdx.z;
  const float* ctl = control + (size_t)b * CPD * FRAMES + (f0 + m16);

  v8f acc0 = {}, acc1 = {};
  for (int k0 = 0; k0 < CPD; k0 += 32) {
    v16bf a;
#pragma unroll
    for (int j = 0; j < 8; ++j)
      a[j] = (bf16_t)ctl[(size_t)(k0 + hf * 8 + j) * FRAMES];
#pragma unroll
    for (int j = 0; j < 8; ++j)
      a[8 + j] = (bf16_t)ctl[(size_t)(k0 + 16 + hf * 8 + j) * FRAMES];
    v16bf b0 = load_b_col(projT, CPD, n0, k0, lane);
    v16bf b1 = load_b_col(projT, CPD, n0 + 16, k0, lane);
    acc0 = wmma_bf(a, b0, acc0);
    acc1 = wmma_bf(a, b1, acc1);
  }
#pragma unroll
  for (int r = 0; r < 8; ++r) {
    size_t row = (size_t)b * FRAMES + f0 + hf * 8 + r;
    x[row * IDIM + n0 + m16]      = (bf16_t)acc0[r];
    x[row * IDIM + n0 + 16 + m16] = (bf16_t)acc1[r];
  }
}

// ---------------------------------------------------------------------------
// 3) xb = x @ input, xd = x @ direct  (M=65536, N=1024, K=1024)
//    A panel async-staged in LDS; one A fragment feeds two WMMAs.
// ---------------------------------------------------------------------------
__global__ __launch_bounds__(128) void gemm_xbd_kernel(
    const bf16_t* __restrict__ x, const bf16_t* __restrict__ inputT,
    const bf16_t* __restrict__ directT, bf16_t* __restrict__ xb,
    bf16_t* __restrict__ xd) {
  __shared__ __align__(16) bf16_t As[16 * 1024];  // 32 KB
  int lane = threadIdx.x & 31, w = threadIdx.x >> 5;
  int hf = lane >> 4, m16 = lane & 15;
  int n0   = blockIdx.x * 64 + w * 16;
  int row0 = blockIdx.y * 16;

  for (int c = threadIdx.x; c < 2048; c += 128) {
    int e0 = c * 8;
    async_b128(&As[e0], x + (size_t)(row0 + (e0 >> 10)) * IDIM + (e0 & 1023));
  }
  WAIT_ASYNC0();
  __syncthreads();

  v8f acc0 = {}, acc1 = {};
  const bf16_t* arow = &As[m16 * 1024];
  for (int k0 = 0; k0 < IDIM; k0 += 32) {
    v16bf a  = load_a_row(arow, k0, hf);
    v16bf b0 = load_b_col(inputT, IDIM, n0, k0, lane);
    v16bf b1 = load_b_col(directT, IDIM, n0, k0, lane);
    acc0 = wmma_bf(a, b0, acc0);
    acc1 = wmma_bf(a, b1, acc1);
  }
#pragma unroll
  for (int r = 0; r < 8; ++r) {
    size_t off = (size_t)(row0 + hf * 8 + r) * SDIM + n0 + m16;
    xb[off] = (bf16_t)acc0[r];
    xd[off] = (bf16_t)acc1[r];
  }
}

// ---------------------------------------------------------------------------
// 4) Sequential scan: s2_t = s @ stateT; s = s2 + xb_t.  s2_t dumped to
//    global; output GEMM deferred (fully parallel).  Per-step xb frame
//    streamed into LDS by the Tensor Data Mover (one tensor_load_to_lds
//    issued by wave 0; 2D tile 16x1024 bf16, rows strided FRAMES*SDIM),
//    overlapped with the phase-1 WMMA work, completed via s_wait_tensorcnt.
// ---------------------------------------------------------------------------
__global__ __launch_bounds__(1024) void scan_kernel(
    const bf16_t* __restrict__ xb, const bf16_t* __restrict__ stateT,
    bf16_t* __restrict__ s2buf) {
  __shared__ __align__(16) bf16_t scb[16 * 1024];  // state s
  __shared__ __align__(16) bf16_t s2b[16 * 1024];  // s @ A
  __shared__ __align__(16) bf16_t xbb[16 * 1024];  // current xb frame (TDM dst)
  int tid = threadIdx.x;
  int lane = tid & 31, w = tid >> 5;
  int hf = lane >> 4, m16 = lane & 15;
  int b0 = blockIdx.x * 16;

  for (int e = tid * 8; e < 16 * 1024; e += 1024 * 8) {
    *(v8bf*)&scb[e] = (v8bf)(bf16_t)0.0f;
  }
  __syncthreads();

  for (int t = 0; t < FRAMES; ++t) {
    // TDM: fetch frame t of xb (16 batch rows x 1024 cols) into LDS.
    if (w == 0) {
      tdm_load_2d_bf16((unsigned)(uintptr_t)&xbb[0],
                       xb + ((size_t)b0 * FRAMES + (size_t)t) * SDIM,
                       /*rows=*/16, /*cols=*/SDIM,
                       /*row_stride=*/(unsigned long long)FRAMES * SDIM);
    }

    // phase 1: s2 = s @ stateT ; 64 n-tiles, 2 per wave, shared A fragment
    {
      int n0 = w * 32;
      v8f acc0 = {}, acc1 = {};
      const bf16_t* arow = &scb[m16 * 1024];
      for (int k0 = 0; k0 < SDIM; k0 += 32) {
        v16bf a  = load_a_row(arow, k0, hf);
        v16bf b0 = load_b_col(stateT, SDIM, n0, k0, lane);
        v16bf b1 = load_b_col(stateT, SDIM, n0 + 16, k0, lane);
        acc0 = wmma_bf(a, b0, acc0);
        acc1 = wmma_bf(a, b1, acc1);
      }
#pragma unroll
      for (int r = 0; r < 8; ++r) {
        s2b[(hf * 8 + r) * 1024 + n0 + m16]      = (bf16_t)acc0[r];
        s2b[(hf * 8 + r) * 1024 + n0 + 16 + m16] = (bf16_t)acc1[r];
      }
    }
    if (w == 0) WAIT_TENSOR0();   // TDM done before anyone reads xbb
    __syncthreads();

    // phase 2: dump s2_t to global; s = s2 + xb_t
    {
      int e0 = tid * 16;
      int mr = e0 >> 10, col = e0 & 1023;
      bf16_t* gdst = s2buf + ((size_t)(b0 + mr) * FRAMES + t) * SDIM + col;
      v8bf s2lo = *(const v8bf*)&s2b[e0];
      v8bf s2hi = *(const v8bf*)&s2b[e0 + 8];
      *(v8bf*)(gdst)     = s2lo;
      *(v8bf*)(gdst + 8) = s2hi;
      v8bf xlo = *(const v8bf*)&xbb[e0];
      v8bf xhi = *(const v8bf*)&xbb[e0 + 8];
      v8bf nlo, nhi;
#pragma unroll
      for (int j = 0; j < 8; ++j) {
        nlo[j] = (bf16_t)((float)s2lo[j] + (float)xlo[j]);
        nhi[j] = (bf16_t)((float)s2hi[j] + (float)xhi[j]);
      }
      *(v8bf*)&scb[e0]     = nlo;
      *(v8bf*)&scb[e0 + 8] = nhi;
    }
    __syncthreads();
  }
}

// ---------------------------------------------------------------------------
// 5) out = s2_all @ outputT + xd  (in place into io) — fully parallel.
// ---------------------------------------------------------------------------
__global__ __launch_bounds__(128) void gemm_out_kernel(
    const bf16_t* __restrict__ s2all, const bf16_t* __restrict__ outputT,
    bf16_t* __restrict__ io) {
  __shared__ __align__(16) bf16_t As[16 * 1024];
  int lane = threadIdx.x & 31, w = threadIdx.x >> 5;
  int hf = lane >> 4, m16 = lane & 15;
  int n0   = blockIdx.x * 64 + w * 16;
  int row0 = blockIdx.y * 16;

  for (int c = threadIdx.x; c < 2048; c += 128) {
    int e0 = c * 8;
    async_b128(&As[e0],
               s2all + (size_t)(row0 + (e0 >> 10)) * SDIM + (e0 & 1023));
  }
  WAIT_ASYNC0();
  __syncthreads();

  v8f acc;
#pragma unroll
  for (int r = 0; r < 8; ++r)
    acc[r] = (float)io[(size_t)(row0 + hf * 8 + r) * IDIM + n0 + m16];

  const bf16_t* arow = &As[m16 * 1024];
  for (int k0 = 0; k0 < SDIM; k0 += 32) {
    v16bf a = load_a_row(arow, k0, hf);
    v16bf b = load_b_col(outputT, SDIM, n0, k0, lane);
    acc = wmma_bf(a, b, acc);
  }
#pragma unroll
  for (int r = 0; r < 8; ++r)
    io[(size_t)(row0 + hf * 8 + r) * IDIM + n0 + m16] = (bf16_t)acc[r];
}

// ---------------------------------------------------------------------------
// 6) Hann window + overlap-add epilogue -> f32 signal
// ---------------------------------------------------------------------------
__global__ void epilogue_kernel(const bf16_t* __restrict__ ob,
                                float* __restrict__ out) {
  size_t idx = (size_t)blockIdx.x * blockDim.x + threadIdx.x;  // B*F*HOP
  int i = (int)(idx & (HOP - 1));
  size_t bf = idx >> 9;  // b*FRAMES + f
  int f = (int)(bf & (FRAMES - 1));
  const float PI2 = 6.283185307179586f;
  float h1 = 0.5f * (1.0f - cosf(PI2 * (float)i / (float)IDIM));
  float v = h1 * (float)ob[bf * IDIM + i];
  if (f > 0) {
    float h2 = 0.5f * (1.0f - cosf(PI2 * (float)(i + HOP) / (float)IDIM));
    v += h2 * (float)ob[(bf - 1) * IDIM + HOP + i];
  }
  out[idx] = v;
}

// ---------------------------------------------------------------------------
// Launch
// ---------------------------------------------------------------------------
extern "C" void kernel_launch(void* const* d_in, const int* in_sizes, int n_in,
                              void* d_out, int out_size, void* d_ws,
                              size_t ws_size, hipStream_t stream) {
  (void)in_sizes; (void)n_in; (void)out_size; (void)ws_size;
  const float* control = (const float*)d_in[0];
  const float* proj    = (const float*)d_in[1];
  const float* stateM  = (const float*)d_in[2];
  const float* inputM  = (const float*)d_in[3];
  const float* outputM = (const float*)d_in[4];
  const float* directM = (const float*)d_in[5];

  char* ws = (char*)d_ws;
  size_t off = 0;
  bf16_t* projT   = (bf16_t*)(ws + off); off += (size_t)CPD  * IDIM * 2;
  bf16_t* stateT  = (bf16_t*)(ws + off); off += (size_t)SDIM * SDIM * 2;
  bf16_t* inputT  = (bf16_t*)(ws + off); off += (size_t)IDIM * SDIM * 2;
  bf16_t* outputT = (bf16_t*)(ws + off); off += (size_t)SDIM * IDIM * 2;
  bf16_t* directT = (bf16_t*)(ws + off); off += (size_t)IDIM * IDIM * 2;
  bf16_t* xbuf    = (bf16_t*)(ws + off); off += (size_t)BATCH * FRAMES * IDIM * 2;
  bf16_t* xbbuf   = (bf16_t*)(ws + off); off += (size_t)BATCH * FRAMES * SDIM * 2;
  bf16_t* xdout   = (bf16_t*)(ws + off); off += (size_t)BATCH * FRAMES * IDIM * 2;
  bf16_t* s2buf   = xbuf;  // x is dead after gemm_xbd; reuse for s2 dump

  dim3 tb(16, 16);
  transpose_to_bf16<<<dim3(IDIM / 16, CPD / 16),  tb, 0, stream>>>(proj,    projT,   CPD,  IDIM);
  transpose_to_bf16<<<dim3(SDIM / 16, SDIM / 16), tb, 0, stream>>>(stateM,  stateT,  SDIM, SDIM);
  transpose_to_bf16<<<dim3(SDIM / 16, IDIM / 16), tb, 0, stream>>>(inputM,  inputT,  IDIM, SDIM);
  transpose_to_bf16<<<dim3(IDIM / 16, SDIM / 16), tb, 0, stream>>>(outputM, outputT, SDIM, IDIM);
  transpose_to_bf16<<<dim3(IDIM / 16, IDIM / 16), tb, 0, stream>>>(directM, directT, IDIM, IDIM);

  gemm_x_kernel<<<dim3(IDIM / 128, FRAMES / 16, BATCH), 128, 0, stream>>>(
      control, projT, xbuf);

  gemm_xbd_kernel<<<dim3(SDIM / 64, (BATCH * FRAMES) / 16), 128, 0, stream>>>(
      xbuf, inputT, directT, xbbuf, xdout);

  scan_kernel<<<BATCH / 16, 1024, 0, stream>>>(xbbuf, stateT, s2buf);

  gemm_out_kernel<<<dim3(IDIM / 64, (BATCH * FRAMES) / 16), 128, 0, stream>>>(
      s2buf, outputT, xdout);

  epilogue_kernel<<<(BATCH * FRAMES * HOP) / 256, 256, 0, stream>>>(
      xdout, (float*)d_out);
}
